// LinearAttention_37349035606743
// MI455X (gfx1250) — compile-verified, matrix operand order checked by reading, not run on hardware
//
#include <hip/hip_runtime.h>
#include <hip/hip_bf16.h>

// ---------------------------------------------------------------------------
// Linear attention for MI455X (gfx1250, wave32, WMMA).
// Heavy math through v_wmma_f32_16x16x32_bf16; weight tiles staged through
// LDS (async-to-LDS when the toolchain exposes it); ping-pong pipelined loads.
// ---------------------------------------------------------------------------

typedef __attribute__((ext_vector_type(16))) __bf16 v16bf;
typedef __attribute__((ext_vector_type(8)))  __bf16 v8bf;
typedef __attribute__((ext_vector_type(16))) float  v16f;
typedef __attribute__((ext_vector_type(8)))  float  v8f;
typedef __attribute__((ext_vector_type(4)))  int    v4i_;

#define BATCH 4
#define SEQ   4096
#define DIM   1024
#define NH    16
#define HD    64
#define MTOT  (BATCH * SEQ)
#define EPSV  1e-6f

#define KC 64                      // K-chunk staged in LDS per step
#define BROW (KC + 16)             // padded LDS row (elems): 160B stride, 32B aligned

#if __has_builtin(__builtin_amdgcn_global_load_async_to_lds_b128)
#define HAVE_ASYNC_LDS 1
#pragma message("CDNA5: async global->LDS path ENABLED")
typedef __attribute__((address_space(1))) v4i_* GAsy;   // global v4i*
typedef __attribute__((address_space(3))) v4i_* LAsy;   // LDS v4i* (32-bit)
#else
#define HAVE_ASYNC_LDS 0
#pragma message("CDNA5: async global->LDS path DISABLED (ds_store fallback)")
#endif

__device__ __forceinline__ void wait_async_zero() {
#if HAVE_ASYNC_LDS
#if __has_builtin(__builtin_amdgcn_s_wait_asynccnt)
  __builtin_amdgcn_s_wait_asynccnt(0);
#else
  asm volatile("s_wait_asynccnt 0x0" ::: "memory");
#endif
#endif
}

__device__ __forceinline__ v8f wmma_bf16(v16bf a, v16bf b, v8f c) {
  return __builtin_amdgcn_wmma_f32_16x16x32_bf16(false, a, false, b, (short)0, c,
                                                 false, false);
}

__device__ __forceinline__ float featmap(float x) {
  // elu(x)+1 == x+1 for x>0, exp(x) for x<=0
  return x > 0.f ? x + 1.f : __expf(x);
}

// ---------------------------------------------------------------------------
// fp32 -> bf16 convert (weights)
// ---------------------------------------------------------------------------
__global__ __launch_bounds__(256) void cvt_f32_bf16(const float* __restrict__ in,
                                                    __bf16* __restrict__ out, int n) {
  int i = blockIdx.x * 256 + threadIdx.x;
  if (i < n) out[i] = (__bf16)in[i];
}

// ---------------------------------------------------------------------------
// Projection GEMM: Y[m,n] = sum_k A[m,k] * W[n,k] + bias[n]
// MODE 0: featuremap, bf16 Q-head layout [B,H,S,HD]
// MODE 1: featuremap, bf16 Kt layout     [B,H,HD,S]
// MODE 2: plain,      bf16 Vt layout     [B,H,HD,S]
// MODE 3: plain,      fp32 [MTOT, DIM]   (final output)
// Block: 256 thr = 8 waves; wave tile 32(M)x64(N); block tile 128x128.
// Weight tile staged in LDS, double-buffered in 128xKC chunks.
// ---------------------------------------------------------------------------
template <int MODE, bool ABF16>
__global__ __launch_bounds__(256) void proj_gemm(const void* __restrict__ Av,
                                                 const __bf16* __restrict__ Bw,
                                                 const float* __restrict__ bias,
                                                 void* __restrict__ Out) {
  __shared__ __bf16 Bs[2][128 * BROW];

  const int lane  = threadIdx.x & 31;
  const int wave  = threadIdx.x >> 5;
  const int waveM = wave & 3;
  const int waveN = wave >> 2;
  const int blockM = blockIdx.y * 128;
  const int blockN = blockIdx.x * 128;

  const int rsel  = lane & 15;
  const int kHalf = (lane >> 4) << 4;
  const int aRow0 = blockM + waveM * 32 + rsel;

  // cooperative fill of one LDS chunk: 128 rows x KC cols, 16B per lane-op
  auto fillB = [&](int buf, int k0) {
#pragma unroll
    for (int t = 0; t < 4; ++t) {
      int seg = threadIdx.x + t * 256;        // 0..1023
      int row = seg >> 3;                      // 0..127
      int c8  = (seg & 7) << 3;                // 0,8,..,56
      const __bf16* src = Bw + (size_t)(blockN + row) * DIM + k0 + c8;
      __bf16* dst = &Bs[buf][row * BROW + c8];
#if HAVE_ASYNC_LDS
      __builtin_amdgcn_global_load_async_to_lds_b128(
          (GAsy)(uintptr_t)src, (LAsy)(unsigned)(uintptr_t)dst, 0, 0);
#else
      *(v8bf*)dst = *(const v8bf*)src;
#endif
    }
  };

  v8f acc[2][4];
#pragma unroll
  for (int i = 0; i < 2; ++i)
#pragma unroll
    for (int j = 0; j < 4; ++j) acc[i][j] = (v8f){0.f, 0.f, 0.f, 0.f, 0.f, 0.f, 0.f, 0.f};

  const float*  Af = (const float*)Av;
  const __bf16* Ab = (const __bf16*)Av;

  fillB(0, 0);
  int buf = 0;

  for (int kc = 0; kc < DIM; kc += KC) {
    wait_async_zero();
    __syncthreads();
    if (kc + KC < DIM) {
      fillB(buf ^ 1, kc + KC);
#pragma unroll
      for (int i = 0; i < 2; ++i)
        __builtin_prefetch(Af + (size_t)(aRow0 + i * 16) * DIM + kc + KC + kHalf, 0, 1);
    }

    // A fragments for both k-substeps of this chunk (issued up front)
    v16bf a[2][2];
#pragma unroll
    for (int ks = 0; ks < 2; ++ks) {
#pragma unroll
      for (int i = 0; i < 2; ++i) {
        if (ABF16) {
          a[ks][i] = *(const v16bf*)(Ab + (size_t)(aRow0 + i * 16) * DIM + kc +
                                     ks * 32 + kHalf);
        } else {
          v16f t = *(const v16f*)(Af + (size_t)(aRow0 + i * 16) * DIM + kc +
                                  ks * 32 + kHalf);
#pragma unroll
          for (int e = 0; e < 16; ++e) a[ks][i][e] = (__bf16)t[e];
        }
      }
    }

#pragma unroll
    for (int ks = 0; ks < 2; ++ks) {
      v16bf b[4];
#pragma unroll
      for (int j = 0; j < 4; ++j)
        b[j] = *(const v16bf*)(&Bs[buf][(waveN * 64 + j * 16 + rsel) * BROW +
                                        ks * 32 + kHalf]);
#pragma unroll
      for (int i = 0; i < 2; ++i)
#pragma unroll
        for (int j = 0; j < 4; ++j) acc[i][j] = wmma_bf16(a[ks][i], b[j], acc[i][j]);
    }
    buf ^= 1;
  }

  // Epilogue. C/D layout: lane L, vgpr e -> row 8*(L/16)+e, col L%16.
  const int mHi = (lane >> 4) << 3;
#pragma unroll
  for (int i = 0; i < 2; ++i) {
#pragma unroll
    for (int j = 0; j < 4; ++j) {
#pragma unroll
      for (int e = 0; e < 8; ++e) {
        int m = blockM + waveM * 32 + i * 16 + mHi + e;
        int n = blockN + waveN * 64 + j * 16 + rsel;
        float v = acc[i][j][e] + bias[n];
        if (MODE <= 1) v = featmap(v);
        if (MODE == 3) {
          ((float*)Out)[(size_t)m * DIM + n] = v;
        } else {
          int b  = m >> 12;
          int s  = m & (SEQ - 1);
          int h  = n >> 6;
          int dk = n & (HD - 1);
          size_t idx;
          if (MODE == 0)
            idx = (((size_t)(b * NH + h) * SEQ) + s) * HD + dk;   // [B,H,S,HD]
          else
            idx = (((size_t)(b * NH + h) * HD) + dk) * SEQ + s;   // [B,H,HD,S]
          ((__bf16*)Out)[idx] = (__bf16)v;
        }
      }
    }
  }
}

// ---------------------------------------------------------------------------
// Ksum[b,h,d] = sum_n Kt[b,h,d,n]
// ---------------------------------------------------------------------------
__global__ __launch_bounds__(256) void ksum_kernel(const __bf16* __restrict__ Kt,
                                                   float* __restrict__ Ksum) {
  int gwave = (blockIdx.x * 256 + threadIdx.x) >> 5;
  int lane  = threadIdx.x & 31;
  const __bf16* row = Kt + (size_t)gwave * SEQ;
  float s = 0.f;
  for (int i = lane * 8; i < SEQ; i += 256) {
    v8bf v = *(const v8bf*)(row + i);
#pragma unroll
    for (int e = 0; e < 8; ++e) s += (float)v[e];
  }
#pragma unroll
  for (int off = 16; off > 0; off >>= 1) s += __shfl_xor(s, off, 32);
  if (lane == 0) Ksum[gwave] = s;
}

// ---------------------------------------------------------------------------
// Zinv[b,h,n] = 1 / (Q[b,h,n,:] . Ksum[b,h,:] + eps)
// ---------------------------------------------------------------------------
__global__ __launch_bounds__(256) void z_kernel(const __bf16* __restrict__ Qh,
                                                const float* __restrict__ Ksum,
                                                float* __restrict__ Zinv) {
  int t = blockIdx.x * 256 + threadIdx.x;
  int bh = t >> 12;
  const __bf16* q = Qh + (size_t)t * HD;
  const float* ks = Ksum + bh * HD;
  float s = 0.f;
#pragma unroll 8
  for (int d = 0; d < HD; ++d) s += (float)q[d] * ks[d];
  Zinv[t] = 1.f / (s + EPSV);
}

// ---------------------------------------------------------------------------
// KV partial: KV[m,d] = sum_n V[n,m] K[n,d] over a 512-chunk of n.
// Fully-unrolled ping-pong (a0/b0 vs a1/b1): no register-copy churn, loads
// for one buffer overlap WMMAs of the other.
// Block: 128 thr = 4 waves; grid (64 heads, 8 chunks).
// ---------------------------------------------------------------------------
__global__ __launch_bounds__(128) void kv_partial(const __bf16* __restrict__ Vt,
                                                  const __bf16* __restrict__ Kt,
                                                  float* __restrict__ KVpart) {
  int bh = blockIdx.x, chunk = blockIdx.y;
  int lane = threadIdx.x & 31, wave = threadIdx.x >> 5;
  const __bf16* Vb = Vt + (size_t)bh * HD * SEQ;
  const __bf16* Kb = Kt + (size_t)bh * HD * SEQ;
  const int rsel  = lane & 15;
  const int kHalf = (lane >> 4) << 4;
  const int mrow  = wave * 16 + rsel;

  v8f acc[4];
#pragma unroll
  for (int j = 0; j < 4; ++j) acc[j] = (v8f){0.f, 0.f, 0.f, 0.f, 0.f, 0.f, 0.f, 0.f};

  const int nStart = chunk * 512;

  v16bf a0, a1, b0[4], b1[4];
  auto loadAB = [&](v16bf& a, v16bf* b, int n0) {
    a = *(const v16bf*)(Vb + (size_t)mrow * SEQ + n0 + kHalf);
#pragma unroll
    for (int j = 0; j < 4; ++j)
      b[j] = *(const v16bf*)(Kb + (size_t)(j * 16 + rsel) * SEQ + n0 + kHalf);
  };

  loadAB(a0, b0, nStart);
  loadAB(a1, b1, nStart + 32);

#pragma unroll
  for (int it = 0; it < 8; ++it) {
#pragma unroll
    for (int j = 0; j < 4; ++j) acc[j] = wmma_bf16(a0, b0[j], acc[j]);
    if (it < 7) loadAB(a0, b0, nStart + it * 64 + 64);
#pragma unroll
    for (int j = 0; j < 4; ++j) acc[j] = wmma_bf16(a1, b1[j], acc[j]);
    if (it < 7) loadAB(a1, b1, nStart + it * 64 + 96);
  }

  float* outp = KVpart + (size_t)(bh * 8 + chunk) * (HD * HD);
  const int mHi = (lane >> 4) << 3;
#pragma unroll
  for (int j = 0; j < 4; ++j)
#pragma unroll
    for (int e = 0; e < 8; ++e)
      outp[(wave * 16 + mHi + e) * HD + j * 16 + rsel] = acc[j][e];
}

__global__ __launch_bounds__(256) void kv_reduce(const float* __restrict__ KVpart,
                                                 __bf16* __restrict__ KV) {
  int bh = blockIdx.x;
  for (int i = threadIdx.x; i < HD * HD; i += 256) {
    float s = 0.f;
#pragma unroll
    for (int c = 0; c < 8; ++c) s += KVpart[(size_t)(bh * 8 + c) * (HD * HD) + i];
    KV[(size_t)bh * (HD * HD) + i] = (__bf16)s;
  }
}

// ---------------------------------------------------------------------------
// out[n,m] = (sum_d Q[n,d] KV[m,d]) * Zinv[n] -> attn bf16 [B,S,D]
// ---------------------------------------------------------------------------
__global__ __launch_bounds__(256) void attn_gemm(const __bf16* __restrict__ Qh,
                                                 const __bf16* __restrict__ KV,
                                                 const float* __restrict__ Zinv,
                                                 __bf16* __restrict__ attn) {
  int bh = blockIdx.y;
  int lane = threadIdx.x & 31, wave = threadIdx.x >> 5;
  int nBase = blockIdx.x * 256 + wave * 32;
  const __bf16* Qb  = Qh + (size_t)bh * SEQ * HD;
  const __bf16* KVb = KV + (size_t)bh * (HD * HD);
  const int rsel  = lane & 15;
  const int kHalf = (lane >> 4) << 4;

  v8f acc[2][4];
#pragma unroll
  for (int i = 0; i < 2; ++i)
#pragma unroll
    for (int j = 0; j < 4; ++j) acc[i][j] = (v8f){0.f, 0.f, 0.f, 0.f, 0.f, 0.f, 0.f, 0.f};

#pragma unroll
  for (int d0 = 0; d0 < HD; d0 += 32) {
    v16bf a[2];
#pragma unroll
    for (int i = 0; i < 2; ++i)
      a[i] = *(const v16bf*)(Qb + (size_t)(nBase + i * 16 + rsel) * HD + d0 + kHalf);
    v16bf b[4];
#pragma unroll
    for (int j = 0; j < 4; ++j)
      b[j] = *(const v16bf*)(KVb + (size_t)(j * 16 + rsel) * HD + d0 + kHalf);
#pragma unroll
    for (int i = 0; i < 2; ++i)
#pragma unroll
      for (int j = 0; j < 4; ++j) acc[i][j] = wmma_bf16(a[i], b[j], acc[i][j]);
  }

  int b_ = bh >> 4, h = bh & (NH - 1);
  const int mHi = (lane >> 4) << 3;
#pragma unroll
  for (int i = 0; i < 2; ++i)
#pragma unroll
    for (int j = 0; j < 4; ++j)
#pragma unroll
      for (int e = 0; e < 8; ++e) {
        int n = nBase + i * 16 + mHi + e;
        int m = j * 16 + rsel;
        float v = acc[i][j][e] * Zinv[(size_t)bh * SEQ + n];
        attn[((size_t)b_ * SEQ + n) * DIM + h * HD + m] = (__bf16)v;
      }
}

// ---------------------------------------------------------------------------
// Launcher
// ---------------------------------------------------------------------------
extern "C" void kernel_launch(void* const* d_in, const int* in_sizes, int n_in,
                              void* d_out, int out_size, void* d_ws, size_t ws_size,
                              hipStream_t stream) {
  (void)in_sizes; (void)n_in; (void)out_size; (void)ws_size;
  const float* query = (const float*)d_in[0];
  const float* key   = (const float*)d_in[1];
  const float* value = (const float*)d_in[2];
  const float* Wq = (const float*)d_in[3];  const float* bq = (const float*)d_in[4];
  const float* Wk = (const float*)d_in[5];  const float* bk = (const float*)d_in[6];
  const float* Wv = (const float*)d_in[7];  const float* bv = (const float*)d_in[8];
  const float* Wo = (const float*)d_in[9];  const float* bo = (const float*)d_in[10];
  float* out = (float*)d_out;

  char*  base = (char*)d_ws;
  size_t off = 0;
  auto carve = [&](size_t bytes) {
    char* p = base + off;
    off = (off + bytes + 255) & ~(size_t)255;
    return p;
  };
  const size_t wElems = (size_t)DIM * DIM;
  __bf16* WqB = (__bf16*)carve(wElems * 2);
  __bf16* WkB = (__bf16*)carve(wElems * 2);
  __bf16* WvB = (__bf16*)carve(wElems * 2);
  __bf16* WoB = (__bf16*)carve(wElems * 2);
  const size_t actElems = (size_t)MTOT * DIM;
  __bf16* Qh   = (__bf16*)carve(actElems * 2);             // [B,H,S,HD]
  __bf16* Kt   = (__bf16*)carve(actElems * 2);             // [B,H,HD,S]
  __bf16* Vt   = (__bf16*)carve(actElems * 2);             // [B,H,HD,S]
  __bf16* attn = (__bf16*)carve(actElems * 2);             // [B,S,D]
  float*  Ksum   = (float*)carve((size_t)BATCH * NH * HD * 4);
  float*  KVpart = (float*)carve((size_t)BATCH * NH * 8 * HD * HD * 4);
  __bf16* KV     = (__bf16*)carve((size_t)BATCH * NH * HD * HD * 2);
  float*  Zinv   = (float*)carve((size_t)BATCH * NH * SEQ * 4);

  dim3 cvtGrid((unsigned)((wElems + 255) / 256));
  cvt_f32_bf16<<<cvtGrid, 256, 0, stream>>>(Wq, WqB, (int)wElems);
  cvt_f32_bf16<<<cvtGrid, 256, 0, stream>>>(Wk, WkB, (int)wElems);
  cvt_f32_bf16<<<cvtGrid, 256, 0, stream>>>(Wv, WvB, (int)wElems);
  cvt_f32_bf16<<<cvtGrid, 256, 0, stream>>>(Wo, WoB, (int)wElems);

  dim3 gGrid(DIM / 128, MTOT / 128);
  proj_gemm<0, false><<<gGrid, 256, 0, stream>>>((const void*)query, WqB, bq, (void*)Qh);
  proj_gemm<1, false><<<gGrid, 256, 0, stream>>>((const void*)key,   WkB, bk, (void*)Kt);
  proj_gemm<2, false><<<gGrid, 256, 0, stream>>>((const void*)value, WvB, bv, (void*)Vt);

  ksum_kernel<<<(BATCH * NH * HD) / 8, 256, 0, stream>>>(Kt, Ksum);
  z_kernel<<<(BATCH * NH * SEQ) / 256, 256, 0, stream>>>(Qh, Ksum, Zinv);

  kv_partial<<<dim3(BATCH * NH, 8), 128, 0, stream>>>(Vt, Kt, KVpart);
  kv_reduce<<<BATCH * NH, 256, 0, stream>>>(KVpart, KV);

  attn_gemm<<<dim3(SEQ / 256, BATCH * NH), 256, 0, stream>>>(Qh, KV, Zinv, attn);

  proj_gemm<3, true><<<gGrid, 256, 0, stream>>>((const void*)attn, WoB, bo, (void*)out);
}
